// QuestionConditionedSelector_87067577024704
// MI455X (gfx1250) — matrix operand order repaired and checked by reading, not compiled, not run on hardware
//
#include <hip/hip_runtime.h>
#include <hip/hip_bf16.h>

// ---------------------------------------------------------------------------
// QuestionConditionedSelector for MI455X (gfx1250, wave32, WMMA + TDM).
// GEMM operands are pre-converted to bf16 (halves HBM stream traffic for the
// panels that dominate: 3x [46080,1024] GEMMs + decoder). LDS tiles are staged
// by the Tensor Data Mover (tensor_load_to_lds, TENSORcnt) and consumed by
// v_wmma_f32_16x16x32_bf16 with fp32 accumulation.
// ---------------------------------------------------------------------------

#define BATCH 16
#define NP    2880
#define DIM   1024
#define SQ    64
#define TD    4096
#define NH    16
#define HD    64
#define KSEL  1152

#define M_VIS (BATCH * NP)    // 46080
#define M_Q   (BATCH * SQ)    // 1024
#define M_SEL (BATCH * KSEL)  // 18432

typedef __attribute__((ext_vector_type(16))) __bf16 v16bf;
typedef __attribute__((ext_vector_type(8)))  float  v8f;

__device__ __forceinline__ __bf16 to_bf16(float f) {
  unsigned u = __builtin_bit_cast(unsigned, f);
  u += 0x7FFFu + ((u >> 16) & 1u);           // round-to-nearest-even
  unsigned short s = (unsigned short)(u >> 16);
  return __builtin_bit_cast(__bf16, s);
}

// ---------------------------------------------------------------------------
// TDM 2D tile load: global (bf16, row-major, row stride `stride0` elements)
// -> LDS at byte offset `lds_off`, tile = tile1 rows x tile0 elements, with
// hardware row padding (pad after 2^(pad_interval+1) DWORDs, pad 1 DWORD).
// D# bitfields per CDNA5 ISA sec. 8.3/8.4.
// ---------------------------------------------------------------------------
#if defined(__AMDGCN__) && __has_builtin(__builtin_amdgcn_tensor_load_to_lds)
#define USE_TDM 1
#else
#define USE_TDM 0
#endif

#if USE_TDM
typedef __attribute__((ext_vector_type(4))) unsigned int tdm_v4u;
typedef __attribute__((ext_vector_type(8))) int          tdm_v8i;
typedef __attribute__((ext_vector_type(4))) int          tdm_v4i;

__device__ __forceinline__ void tdm_load_2d(
    const void* gptr, unsigned lds_off,
    unsigned tile0, unsigned tile1,
    unsigned dim0, unsigned dim1,
    unsigned long long stride0, unsigned pad_interval)
{
  const unsigned long long ga = (unsigned long long)gptr;
  tdm_v4u g0;
  g0.x = 1u;                                   // count=1, user descriptor
  g0.y = lds_off;                              // lds_addr (bytes)
  g0.z = (unsigned)ga;                         // global_addr[31:0]
  g0.w = (unsigned)((ga >> 32) & 0x01FFFFFFull) | (2u << 30);  // addr[56:32], type=2
  const unsigned w0 = (1u << 16)               // data_size = 2 bytes (bf16)
                    | (1u << 20)               // pad_enable
                    | (pad_interval << 22)     // pad interval (2^(v+1) DWORDs)
                    | (0u << 25);              // pad amount = 1 DWORD
  tdm_v8i g1;
  g1[0] = (int)w0;
  g1[1] = (int)((dim0 & 0xFFFFu) << 16);                               // tdim0[15:0]
  g1[2] = (int)(((dim0 >> 16) & 0xFFFFu) | ((dim1 & 0xFFFFu) << 16));  // tdim0 hi, tdim1 lo
  g1[3] = (int)(((dim1 >> 16) & 0xFFFFu) | ((tile0 & 0xFFFFu) << 16)); // tdim1 hi, tile0
  g1[4] = (int)(tile1 & 0xFFFFu);                                      // tile1, tile2=0
  g1[5] = (int)(unsigned)(stride0 & 0xFFFFFFFFull);                    // stride0[31:0]
  g1[6] = (int)(unsigned)((stride0 >> 32) & 0xFFFFull);                // stride0[47:32]
  g1[7] = 0;
  tdm_v4i z4 = {0, 0, 0, 0};
#if __clang_major__ >= 23
  tdm_v8i z8 = {0, 0, 0, 0, 0, 0, 0, 0};
  __builtin_amdgcn_tensor_load_to_lds(g0, g1, z4, z4, z8, 0);
#else
  __builtin_amdgcn_tensor_load_to_lds(g0, g1, z4, z4, 0);
#endif
}
#endif

// ---------------------------------------------------------------------------
// Tiled WMMA GEMM (bf16 in, fp32 accumulate): C = act(A[M,K] @ W[K,N] + bias)
// Block tile 128x128x32, 256 threads = 8 waves, each wave owns 64x32 =
// 4(M) x 2(N) wmma accumulators. Staging via TDM (wave 0 issues descriptors,
// s_wait_tensorcnt, workgroup barrier). ACT: 0=none,1=GELU,2=ReLU.
// OUTB: 0 -> fp32 store to Cf, 1 -> bf16 store to Cb.
// ---------------------------------------------------------------------------
template <int ACT, int OUTB>
__global__ __launch_bounds__(256) void gemm_bf16(
    const __bf16* __restrict__ A, const __bf16* __restrict__ W,
    const float* __restrict__ bias,
    float* __restrict__ Cf, __bf16* __restrict__ Cb,
    int M, int N, int Kd)
{
  __shared__ __bf16 As[128][34];   // [m][k] rows padded by TDM (64B + 4B pad)
  __shared__ __bf16 Bs[32][130];   // [k][n] rows padded by TDM (256B + 4B pad)

  const int t    = threadIdx.x;
  const int bm   = blockIdx.y * 128;
  const int bn   = blockIdx.x * 128;
  const int w    = t >> 5;
  const int lane = t & 31;
  const int lrow = lane & 15;
  const int kh   = lane >> 4;      // lane half selects K-half (contiguous k order used for A and B)
  const int wm   = (w >> 2) * 64;  // 0 / 64
  const int wn   = (w & 3) * 32;   // 0/32/64/96

  v8f acc[4][2];
  const v8f vzero = {0.f,0.f,0.f,0.f,0.f,0.f,0.f,0.f};
  #pragma unroll
  for (int i = 0; i < 4; ++i)
    #pragma unroll
    for (int j = 0; j < 2; ++j) acc[i][j] = vzero;

  for (int kk = 0; kk < Kd; kk += 32) {
#if USE_TDM
    if (w == 0) {                  // one wave issues both tile DMAs
      const unsigned asBase = (unsigned)(unsigned long long)(void*)&As[0][0];
      const unsigned bsBase = (unsigned)(unsigned long long)(void*)&Bs[0][0];
      tdm_load_2d(A + (size_t)bm * Kd + kk, asBase,
                  /*tile0=*/32, /*tile1=*/128,
                  /*dim0=*/(unsigned)(Kd - kk), /*dim1=*/(unsigned)(M - bm),
                  /*stride0=*/(unsigned long long)Kd, /*pad_interval=*/3);
      tdm_load_2d(W + (size_t)kk * N + bn, bsBase,
                  /*tile0=*/128, /*tile1=*/32,
                  /*dim0=*/(unsigned)(N - bn), /*dim1=*/(unsigned)(Kd - kk),
                  /*stride0=*/(unsigned long long)N, /*pad_interval=*/5);
      __builtin_amdgcn_s_wait_tensorcnt(0);
    }
#else
    for (int i = t; i < 128 * 32; i += 256) {
      int r = i >> 5, c = i & 31;
      As[r][c] = A[(size_t)(bm + r) * Kd + (kk + c)];
    }
    for (int i = t; i < 32 * 128; i += 256) {
      int k = i >> 7, n = i & 127;
      Bs[k][n] = W[(size_t)(kk + k) * N + (bn + n)];
    }
#endif
    __syncthreads();

    // A fragments: lane half kh: elements 0..15 hold K=kh*16+0..15
    // (same k order as the B fragment).
    v16bf afr[4];
    #pragma unroll
    for (int mt = 0; mt < 4; ++mt) {
      const __bf16* ap = &As[wm + mt * 16 + lrow][0];
      #pragma unroll
      for (int i = 0; i < 8; ++i) afr[mt][i]     = ap[kh * 16 + i];
      #pragma unroll
      for (int i = 0; i < 8; ++i) afr[mt][8 + i] = ap[kh * 16 + 8 + i];
    }
    #pragma unroll
    for (int nt = 0; nt < 2; ++nt) {
      // B fragment: element i -> K = kh*16 + i, N = lane&15 (Bs is [k][n]).
      v16bf bfr;
      const __bf16* bp = &Bs[kh * 16][wn + nt * 16 + lrow];
      #pragma unroll
      for (int i = 0; i < 16; ++i) bfr[i] = bp[(size_t)i * 130];
      #pragma unroll
      for (int mt = 0; mt < 4; ++mt) {
        acc[mt][nt] = __builtin_amdgcn_wmma_f32_16x16x32_bf16(
            false, afr[mt], false, bfr, (short)0, acc[mt][nt], false, false);
      }
    }
    __syncthreads();
  }

  // Epilogue. C/D layout: VGPR r -> M = r + 8*kh, N = lane&15.
  #pragma unroll
  for (int mt = 0; mt < 4; ++mt) {
    #pragma unroll
    for (int nt = 0; nt < 2; ++nt) {
      int n = bn + wn + nt * 16 + lrow;
      float bv = bias[n];
      #pragma unroll
      for (int r = 0; r < 8; ++r) {
        int m = bm + wm + mt * 16 + kh * 8 + r;
        float val = acc[mt][nt][r] + bv;
        if (ACT == 1) val = 0.5f * val * (1.0f + erff(val * 0.70710678118f));
        else if (ACT == 2) val = fmaxf(val, 0.0f);
        if (OUTB) Cb[(size_t)m * N + n] = to_bf16(val);
        else      Cf[(size_t)m * N + n] = val;
      }
    }
  }
}

// fp32 -> bf16 pre-conversion (amortized: each panel is re-read many times)
__global__ __launch_bounds__(256) void cvt_bf16_kernel(
    const float* __restrict__ in, __bf16* __restrict__ out, long n)
{
  const long i = (long)blockIdx.x * 256 + threadIdx.x;
  if (i < n) out[i] = to_bf16(in[i]);
}

// ---------------------------------------------------------------------------
// LayerNorm over DIM=1024, optional residual; writes fp32 and/or bf16.
// ---------------------------------------------------------------------------
__global__ __launch_bounds__(256) void ln_kernel(
    const float* __restrict__ x, const float* __restrict__ res,
    const float* __restrict__ g, const float* __restrict__ be,
    float* __restrict__ outf, __bf16* __restrict__ outb)
{
  const int r = blockIdx.x, t = threadIdx.x;
  __shared__ float row[DIM];
  __shared__ float red[256];
  const size_t base = (size_t)r * DIM;

  float part = 0.f;
  for (int i = t; i < DIM; i += 256) {
    float v = x[base + i];
    if (res) v += res[base + i];
    row[i] = v;
    part += v;
  }
  red[t] = part; __syncthreads();
  for (int s = 128; s > 0; s >>= 1) { if (t < s) red[t] += red[t + s]; __syncthreads(); }
  const float mean = red[0] * (1.f / DIM);
  __syncthreads();

  float vp = 0.f;
  for (int i = t; i < DIM; i += 256) { float d = row[i] - mean; vp += d * d; }
  red[t] = vp; __syncthreads();
  for (int s = 128; s > 0; s >>= 1) { if (t < s) red[t] += red[t + s]; __syncthreads(); }
  const float inv = rsqrtf(red[0] * (1.f / DIM) + 1e-5f);

  for (int i = t; i < DIM; i += 256) {
    const float v = (row[i] - mean) * inv * g[i] + be[i];
    if (outf) outf[base + i] = v;
    if (outb) outb[base + i] = to_bf16(v);
  }
}

// ---------------------------------------------------------------------------
// Fused cross attention for one (b,q): scores -> per-head softmax ->
// head-averaged attention weights -> ctx (bf16, feeds the wo GEMM).
// ---------------------------------------------------------------------------
__global__ __launch_bounds__(256) void attn_kernel(
    const float* __restrict__ q, const float* __restrict__ k,
    const float* __restrict__ v, __bf16* __restrict__ ctx,
    float* __restrict__ attnw)
{
  const int bq = blockIdx.x;            // 0 .. B*NP-1
  const int b  = bq / NP;
  const int t  = threadIdx.x;
  __shared__ float qrow[DIM];
  __shared__ float sc[NH * SQ];

  for (int i = t; i < DIM; i += 256) qrow[i] = q[(size_t)bq * DIM + i];
  __syncthreads();

  for (int p = t; p < NH * SQ; p += 256) {
    const int h = p >> 6, j = p & 63;
    const float* kr = k + (size_t)(b * SQ + j) * DIM + h * HD;
    const float* qh = qrow + h * HD;
    float s = 0.f;
    #pragma unroll 8
    for (int d = 0; d < HD; ++d) s += qh[d] * kr[d];
    sc[p] = s * 0.125f;                 // 1/sqrt(64)
  }
  __syncthreads();

  if (t < NH) {                          // per-head softmax over 64 keys
    float m = -1e30f;
    for (int j = 0; j < SQ; ++j) m = fmaxf(m, sc[t * SQ + j]);
    float ssum = 0.f;
    for (int j = 0; j < SQ; ++j) { float e = __expf(sc[t * SQ + j] - m); sc[t * SQ + j] = e; ssum += e; }
    const float inv = 1.f / ssum;
    for (int j = 0; j < SQ; ++j) sc[t * SQ + j] *= inv;
  }
  __syncthreads();

  if (t < SQ) {                          // mean over heads (torch MHA behavior)
    float a = 0.f;
    for (int h = 0; h < NH; ++h) a += sc[h * SQ + t];
    attnw[(size_t)bq * SQ + t] = a * (1.f / NH);
  }

  for (int d = t; d < DIM; d += 256) {   // ctx = probs @ v, coalesced over d
    const int h = d >> 6;
    float a = 0.f;
    #pragma unroll 8
    for (int j = 0; j < SQ; ++j) a += sc[h * SQ + j] * v[(size_t)(b * SQ + j) * DIM + d];
    ctx[(size_t)bq * DIM + d] = to_bf16(a);
  }
}

// Importance head: dot(h[row,0:128], iw4) + b -> sigmoid
__global__ __launch_bounds__(128) void head_kernel(
    const float* __restrict__ h, const float* __restrict__ w4,
    const float* __restrict__ b4, float* __restrict__ imp)
{
  const int row = blockIdx.x, t = threadIdx.x;
  __shared__ float red[128];
  red[t] = h[(size_t)row * 128 + t] * w4[t];
  __syncthreads();
  for (int s = 64; s > 0; s >>= 1) { if (t < s) red[t] += red[t + s]; __syncthreads(); }
  if (t == 0) imp[row] = 1.f / (1.f + __expf(-(red[0] + b4[0])));
}

// ---------------------------------------------------------------------------
// Deterministic per-batch top-K via binary search on float bit pattern
// (sigmoid outputs positive -> uint order == float order), serial compaction.
// ---------------------------------------------------------------------------
__global__ __launch_bounds__(256) void topk_kernel(
    const float* __restrict__ imp, int* __restrict__ idxs,
    float* __restrict__ idxf, float* __restrict__ mask)
{
  const int b = blockIdx.x, t = threadIdx.x;
  __shared__ int red[256];
  const float* s = imp + (size_t)b * NP;

  unsigned lo = 0u, hi = 0xFFFFFFFFu;    // invariant: count(>=lo) >= K
  while (lo < hi) {
    const unsigned mid = lo + 1u + ((hi - lo - 1u) >> 1);   // in (lo, hi]
    int c = 0;
    for (int i = t; i < NP; i += 256)
      if (__float_as_uint(s[i]) >= mid) c++;
    red[t] = c; __syncthreads();
    for (int st = 128; st > 0; st >>= 1) { if (t < st) red[t] += red[t + st]; __syncthreads(); }
    const int total = red[0];
    __syncthreads();
    if (total >= KSEL) lo = mid; else hi = mid - 1u;
  }
  const unsigned thr = lo;               // kth-largest bit pattern

  if (t == 0) {                          // deterministic compaction
    int c = 0;
    for (int i = 0; i < NP; ++i)
      if (__float_as_uint(s[i]) > thr) idxs[b * KSEL + c++] = i;
    for (int i = 0; i < NP && c < KSEL; ++i)
      if (__float_as_uint(s[i]) == thr) idxs[b * KSEL + c++] = i;
  }
  __syncthreads();

  for (int i = t; i < NP; i += 256) mask[(size_t)b * NP + i] = 0.f;
  __syncthreads();
  for (int j = t; j < KSEL; j += 256) {
    const int i = idxs[b * KSEL + j];
    mask[(size_t)b * NP + i] = 1.f;
    idxf[(size_t)b * KSEL + j] = (float)i;
  }
}

__global__ __launch_bounds__(256) void gather_kernel(
    const float* __restrict__ vis, const int* __restrict__ idxs,
    float* __restrict__ sel, __bf16* __restrict__ selb)
{
  const size_t i = (size_t)blockIdx.x * 256 + threadIdx.x;   // over B*K*D
  const int d  = (int)(i & (DIM - 1));
  const size_t rk = i / DIM;
  const int b  = (int)(rk / KSEL);
  const int kk = (int)(rk % KSEL);
  const float v = vis[((size_t)b * NP + idxs[b * KSEL + kk]) * DIM + d];
  sel[i]  = v;
  selb[i] = to_bf16(v);
}

__global__ __launch_bounds__(256) void scatter_kernel(
    const float* __restrict__ rp, const int* __restrict__ idxs,
    float* __restrict__ rec)
{
  const size_t i = (size_t)blockIdx.x * 256 + threadIdx.x;   // over B*K*D
  const int d  = (int)(i & (DIM - 1));
  const size_t rk = i / DIM;
  const int b  = (int)(rk / KSEL);
  const int kk = (int)(rk % KSEL);
  rec[((size_t)b * NP + idxs[b * KSEL + kk]) * DIM + d] = rp[i];
}

__global__ __launch_bounds__(256) void fill_kernel(float* __restrict__ p, long n)
{
  const long i = (long)blockIdx.x * 256 + threadIdx.x;
  if (i < n) p[i] = 0.f;
}

// ---------------------------------------------------------------------------
extern "C" void kernel_launch(void* const* d_in, const int* in_sizes, int n_in,
                              void* d_out, int out_size, void* d_ws, size_t ws_size,
                              hipStream_t stream) {
  const float* visual = (const float*)d_in[0];
  const float* qe     = (const float*)d_in[1];
  const float* qp_w1  = (const float*)d_in[2];
  const float* qp_b1  = (const float*)d_in[3];
  const float* qp_w2  = (const float*)d_in[4];
  const float* qp_b2  = (const float*)d_in[5];
  const float* qp_lng = (const float*)d_in[6];
  const float* qp_lnb = (const float*)d_in[7];
  const float* wq = (const float*)d_in[8];  const float* bq = (const float*)d_in[9];
  const float* wk = (const float*)d_in[10]; const float* bk = (const float*)d_in[11];
  const float* wv = (const float*)d_in[12]; const float* bv = (const float*)d_in[13];
  const float* wo = (const float*)d_in[14]; const float* bo = (const float*)d_in[15];
  const float* ca_lng = (const float*)d_in[16];
  const float* ca_lnb = (const float*)d_in[17];
  const float* iw1 = (const float*)d_in[18]; const float* ib1 = (const float*)d_in[19];
  const float* iw2 = (const float*)d_in[20]; const float* ib2 = (const float*)d_in[21];
  const float* iw3 = (const float*)d_in[22]; const float* ib3 = (const float*)d_in[23];
  const float* iw4 = (const float*)d_in[24]; const float* ib4 = (const float*)d_in[25];
  const float* dw1 = (const float*)d_in[26]; const float* db1 = (const float*)d_in[27];
  const float* dw2 = (const float*)d_in[28]; const float* db2 = (const float*)d_in[29];

  // ---- workspace layout (byte offsets) ----
  char* wsb = (char*)d_ws;
  float*  fA   = (float*)wsb;                         // 47,185,920 f32: q -> ctxO -> h3 -> rp
  float*  fS   = (float*)(wsb + 188743680ull);        // qp_lin | kbuf | vbuf
  __bf16* bR1  = (__bf16*)(wsb + 201326592ull);       // 47,185,920 bf16: visual_b -> h1b/h2b -> dech_b
  __bf16* bR2  = (__bf16*)(wsb + 295698432ull);       // 47,185,920 bf16: ctx_b -> cond_b -> sel_b
  __bf16* bSm  = (__bf16*)(wsb + 390070272ull);       // qe_b | qp_h1_b | qp_b
  __bf16* bW   = (__bf16*)(wsb + 404750336ull);       // bf16 weights (19,562,496)
  int*    idxs = (int*)(wsb + 443875328ull);          // [B,K] int

  float* qp_lin = fS;
  float* kbuf   = fS + 1048576ull;
  float* vbuf   = fS + 2097152ull;

  __bf16* visual_b = bR1;
  __bf16* h1b      = bR1;                     // after visual_b is dead
  __bf16* h2b      = bR1 + 23592960ull;
  __bf16* dech_b   = bR1;                     // after h1b/h2b are dead
  __bf16* ctx_b    = bR2;
  __bf16* cond_b   = bR2;                     // after ctx_b is dead
  __bf16* sel_b    = bR2;                     // after cond_b is dead
  __bf16* qe_b     = bSm;
  __bf16* qp_h1_b  = bSm + 4194304ull;
  __bf16* qp_b     = bSm + 6291456ull;

  __bf16* qp_w1_b = bW;
  __bf16* qp_w2_b = bW + 8388608ull;
  __bf16* wq_b    = bW + 10485760ull;
  __bf16* wk_b    = bW + 11534336ull;
  __bf16* wv_b    = bW + 12582912ull;
  __bf16* wo_b    = bW + 13631488ull;
  __bf16* iw1_b   = bW + 14680064ull;
  __bf16* iw2_b   = bW + 15204352ull;
  __bf16* iw3_b   = bW + 15335424ull;
  __bf16* dw1_b   = bW + 15368192ull;
  __bf16* dw2_b   = bW + 17465344ull;

  // ---- output layout (floats, tuple order) ----
  float* out      = (float*)d_out;
  float* out_sel  = out;                  // [16,1152,1024]
  float* out_imp  = out + 18874368ull;    // [16,2880,1]
  float* out_mask = out + 18920448ull;    // [16,2880,1]
  float* out_rec  = out + 18966528ull;    // [16,2880,1024]
  float* out_idx  = out + 66152448ull;    // [16,1152]
  float* out_attn = out + 66170880ull;    // [16,2880,64]

  const dim3 blk(256);

  // 0) one-time fp32 -> bf16 conversion of GEMM operands
  cvt_bf16_kernel<<<16384,  blk, 0, stream>>>(qe,     qe_b,     4194304L);
  cvt_bf16_kernel<<<184320, blk, 0, stream>>>(visual, visual_b, 47185920L);
  cvt_bf16_kernel<<<32768,  blk, 0, stream>>>(qp_w1,  qp_w1_b,  8388608L);
  cvt_bf16_kernel<<<8192,   blk, 0, stream>>>(qp_w2,  qp_w2_b,  2097152L);
  cvt_bf16_kernel<<<4096,   blk, 0, stream>>>(wq,     wq_b,     1048576L);
  cvt_bf16_kernel<<<4096,   blk, 0, stream>>>(wk,     wk_b,     1048576L);
  cvt_bf16_kernel<<<4096,   blk, 0, stream>>>(wv,     wv_b,     1048576L);
  cvt_bf16_kernel<<<4096,   blk, 0, stream>>>(wo,     wo_b,     1048576L);
  cvt_bf16_kernel<<<2048,   blk, 0, stream>>>(iw1,    iw1_b,    524288L);
  cvt_bf16_kernel<<<512,    blk, 0, stream>>>(iw2,    iw2_b,    131072L);
  cvt_bf16_kernel<<<128,    blk, 0, stream>>>(iw3,    iw3_b,    32768L);
  cvt_bf16_kernel<<<8192,   blk, 0, stream>>>(dw1,    dw1_b,    2097152L);
  cvt_bf16_kernel<<<8192,   blk, 0, stream>>>(dw2,    dw2_b,    2097152L);

  // 1) Question projector: Linear+GELU, Linear, LayerNorm
  gemm_bf16<1,1><<<dim3(16, 8),  blk, 0, stream>>>(qe_b, qp_w1_b, qp_b1, nullptr, qp_h1_b, M_Q, 2 * DIM, TD);
  gemm_bf16<0,0><<<dim3(8, 8),   blk, 0, stream>>>(qp_h1_b, qp_w2_b, qp_b2, qp_lin, nullptr, M_Q, DIM, 2 * DIM);
  ln_kernel<<<M_Q, blk, 0, stream>>>(qp_lin, nullptr, qp_lng, qp_lnb, nullptr, qp_b);

  // 2) Q/K/V projections
  gemm_bf16<0,0><<<dim3(8, 360), blk, 0, stream>>>(visual_b, wq_b, bq, fA,   nullptr, M_VIS, DIM, DIM);
  gemm_bf16<0,0><<<dim3(8, 8),   blk, 0, stream>>>(qp_b,     wk_b, bk, kbuf, nullptr, M_Q,   DIM, DIM);
  gemm_bf16<0,0><<<dim3(8, 8),   blk, 0, stream>>>(qp_b,     wv_b, bv, vbuf, nullptr, M_Q,   DIM, DIM);

  // 3) Fused attention -> ctx_b (bf16) + head-averaged attention weights
  attn_kernel<<<M_VIS, blk, 0, stream>>>(fA, kbuf, vbuf, ctx_b, out_attn);

  // 4) Output projection + residual LayerNorm -> conditioned (bf16)
  gemm_bf16<0,0><<<dim3(8, 360), blk, 0, stream>>>(ctx_b, wo_b, bo, fA, nullptr, M_VIS, DIM, DIM);
  ln_kernel<<<M_VIS, blk, 0, stream>>>(fA, visual, ca_lng, ca_lnb, nullptr, cond_b);

  // 5) Importance MLP (ReLU x3) + sigmoid head
  gemm_bf16<2,1><<<dim3(4, 360), blk, 0, stream>>>(cond_b, iw1_b, ib1, nullptr, h1b, M_VIS, 512, DIM);
  gemm_bf16<2,1><<<dim3(2, 360), blk, 0, stream>>>(h1b, iw2_b, ib2, nullptr, h2b, M_VIS, 256, 512);
  gemm_bf16<2,0><<<dim3(1, 360), blk, 0, stream>>>(h2b, iw3_b, ib3, fA, nullptr, M_VIS, 128, 256);
  head_kernel<<<M_VIS, dim3(128), 0, stream>>>(fA, iw4, ib4, out_imp);

  // 6) Top-K selection + mask/idx, gather selected features (fp32 + bf16)
  topk_kernel<<<BATCH, blk, 0, stream>>>(out_imp, idxs, out_idx, out_mask);
  gather_kernel<<<73728, blk, 0, stream>>>(visual, idxs, out_sel, sel_b);

  // 7) Reconstruction decoder + scatter back to full resolution
  gemm_bf16<2,1><<<dim3(16, 144), blk, 0, stream>>>(sel_b, dw1_b, db1, nullptr, dech_b, M_SEL, 2 * DIM, DIM);
  gemm_bf16<0,0><<<dim3(8, 144),  blk, 0, stream>>>(dech_b, dw2_b, db2, fA, nullptr, M_SEL, DIM, 2 * DIM);
  fill_kernel<<<184320, blk, 0, stream>>>(out_rec, 47185920L);
  scatter_kernel<<<73728, blk, 0, stream>>>(fA, idxs, out_rec);
}
